// RadialBasisFunctionKernel_58720792871618
// MI455X (gfx1250) — matrix operand order repaired and checked by reading, not compiled, not run on hardware
//
#include <hip/hip_runtime.h>
#include <hip/hip_bf16.h>

typedef __attribute__((ext_vector_type(16))) __bf16 v16bf;
typedef __attribute__((ext_vector_type(4)))  __bf16 v4bf;
typedef __attribute__((ext_vector_type(8)))  float  v8f;

#define D_DIM      64
#define TILE_N     128   // rows of A per workgroup
#define TILE_M     64    // rows of B per workgroup
#define LDS_STRIDE 72    // bf16 elems per LDS row (64 data + 8 pad for bank spread)

union Frag { v16bf v; uint4 q[2]; };

// A-matrix fragment (16x32 bf16): lane l holds M=l&15; elems 0-15 at K=k0..k0+15 (contiguous)
__device__ __forceinline__ v16bf load_a_frag(const __bf16* base, int row, int k0) {
  const __bf16* p = base + row * LDS_STRIDE + k0;
  Frag u;
  u.q[0] = *(const uint4*)(p);        // 8 bf16
  u.q[1] = *(const uint4*)(p + 8);    // 8 bf16, +8 elements
  return u.v;
}
// B-matrix fragment (32x16 bf16): lane l holds N=l&15; elems 0-15 at K=k0..k0+15 (contiguous)
__device__ __forceinline__ v16bf load_b_frag(const __bf16* base, int row, int k0) {
  const __bf16* p = base + row * LDS_STRIDE + k0;
  Frag u;
  u.q[0] = *(const uint4*)(p);
  u.q[1] = *(const uint4*)(p + 8);
  return u.v;
}

__global__ __launch_bounds__(256)
void rbf_wmma_kernel(const float* __restrict__ A, const float* __restrict__ B,
                     float* __restrict__ out, int N, int M)
{
  __shared__ __bf16 Ahi[TILE_N * LDS_STRIDE];
  __shared__ __bf16 Alo[TILE_N * LDS_STRIDE];
  __shared__ __bf16 Bhi[TILE_M * LDS_STRIDE];
  __shared__ __bf16 Blo[TILE_M * LDS_STRIDE];
  __shared__ float  normA[TILE_N];
  __shared__ float  normB[TILE_M];

  const int t = threadIdx.x;
  const int rowBase = blockIdx.y * TILE_N;
  const int colBase = blockIdx.x * TILE_M;

  // ---- Stage A tile: 128 rows x 16 float4 = 2048 float4, 256 thr x 8 iters ----
#pragma unroll
  for (int i = 0; i < 8; ++i) {
    int idx = i * 256 + t;
    int r = idx >> 4, c4 = idx & 15;
    float4 a = *(const float4*)(A + (size_t)(rowBase + r) * D_DIM + c4 * 4);
    float av[4] = {a.x, a.y, a.z, a.w};
    v4bf hi, lo;
#pragma unroll
    for (int j = 0; j < 4; ++j) {
      __bf16 h = (__bf16)av[j];
      hi[j] = h;
      lo[j] = (__bf16)(av[j] - (float)h);
    }
    *(v4bf*)(Ahi + r * LDS_STRIDE + c4 * 4) = hi;
    *(v4bf*)(Alo + r * LDS_STRIDE + c4 * 4) = lo;
  }
  // ---- Stage B tile: 64 rows x 16 float4 = 1024 float4, 4 iters ----
#pragma unroll
  for (int i = 0; i < 4; ++i) {
    int idx = i * 256 + t;
    int r = idx >> 4, c4 = idx & 15;
    float4 b = *(const float4*)(B + (size_t)(colBase + r) * D_DIM + c4 * 4);
    float bv[4] = {b.x, b.y, b.z, b.w};
    v4bf hi, lo;
#pragma unroll
    for (int j = 0; j < 4; ++j) {
      __bf16 h = (__bf16)bv[j];
      hi[j] = h;
      lo[j] = (__bf16)(bv[j] - (float)h);
    }
    *(v4bf*)(Bhi + r * LDS_STRIDE + c4 * 4) = hi;
    *(v4bf*)(Blo + r * LDS_STRIDE + c4 * 4) = lo;
  }
  // ---- Row norms in fp32 (inputs are L2-hot) ----
  if (t < TILE_N) {
    const float* src = A + (size_t)(rowBase + t) * D_DIM;
    float s = 0.f;
#pragma unroll
    for (int j = 0; j < 16; ++j) {
      float4 v = *(const float4*)(src + j * 4);
      s = fmaf(v.x, v.x, s); s = fmaf(v.y, v.y, s);
      s = fmaf(v.z, v.z, s); s = fmaf(v.w, v.w, s);
    }
    normA[t] = s;
  } else if (t < TILE_N + TILE_M) {
    const float* src = B + (size_t)(colBase + (t - TILE_N)) * D_DIM;
    float s = 0.f;
#pragma unroll
    for (int j = 0; j < 16; ++j) {
      float4 v = *(const float4*)(src + j * 4);
      s = fmaf(v.x, v.x, s); s = fmaf(v.y, v.y, s);
      s = fmaf(v.z, v.z, s); s = fmaf(v.w, v.w, s);
    }
    normB[t - TILE_N] = s;
  }
  __syncthreads();

  // ---- Wave-level 32x32 sub-block: 2x2 tiles of 16x16 ----
  const int lane = t & 31, wave = t >> 5;
  const int half = lane >> 4, l16 = lane & 15;
  const int wr = wave >> 1;   // 0..3 : row block of 32
  const int wc = wave & 1;    // 0..1 : col block of 32

  const v8f vzero = {0.f,0.f,0.f,0.f,0.f,0.f,0.f,0.f};
  v8f acc[2][2];
#pragma unroll
  for (int rt = 0; rt < 2; ++rt)
#pragma unroll
    for (int ct = 0; ct < 2; ++ct) acc[rt][ct] = vzero;

  // bf16x3 split: cross = hi*hi + hi*lo + lo*hi  (fp32 accumulate)
#pragma unroll
  for (int pass = 0; pass < 3; ++pass) {
    const __bf16* As = (pass < 2) ? Ahi : Alo;
    const __bf16* Bs = (pass == 1) ? Blo : Bhi;
#pragma unroll
    for (int kb = 0; kb < D_DIM; kb += 32) {
      v16bf af[2], bfr[2];
#pragma unroll
      for (int rt = 0; rt < 2; ++rt)
        af[rt] = load_a_frag(As, wr * 32 + rt * 16 + l16, kb + half * 16);
#pragma unroll
      for (int ct = 0; ct < 2; ++ct)
        bfr[ct] = load_b_frag(Bs, wc * 32 + ct * 16 + l16, kb + half * 16);
#pragma unroll
      for (int rt = 0; rt < 2; ++rt)
#pragma unroll
        for (int ct = 0; ct < 2; ++ct)
          acc[rt][ct] = __builtin_amdgcn_wmma_f32_16x16x32_bf16(
              false, af[rt], false, bfr[ct], (short)0, acc[rt][ct], false, false);
    }
  }

  // ---- Epilogue: exp(-0.5 * max(0, |a|^2 + |b|^2 - 2*cross)) ----
  // C layout: VGPR e, lanes 0-15 -> M=e, N=lane; lanes 16-31 -> M=e+8, N=lane-16
  // Output is a 256MB stream (> 192MB L2): use non-temporal stores so the
  // write stream doesn't evict the L2-resident input rows re-read by other tiles.
#pragma unroll
  for (int rt = 0; rt < 2; ++rt) {
#pragma unroll
    for (int ct = 0; ct < 2; ++ct) {
      const int ml = wc * 32 + ct * 16 + l16;
      const float sqm = normB[ml];
#pragma unroll
      for (int e = 0; e < 8; ++e) {
        const int nl = wr * 32 + rt * 16 + e + 8 * half;
        float sqd = normA[nl] + sqm - 2.0f * acc[rt][ct][e];
        sqd = fmaxf(sqd, 0.0f);
        float val = __expf(-0.5f * sqd);
        __builtin_nontemporal_store(
            val, out + (size_t)(rowBase + nl) * (size_t)M + (size_t)(colBase + ml));
      }
    }
  }
}

extern "C" void kernel_launch(void* const* d_in, const int* in_sizes, int n_in,
                              void* d_out, int out_size, void* d_ws, size_t ws_size,
                              hipStream_t stream) {
  (void)n_in; (void)out_size; (void)d_ws; (void)ws_size;
  const float* A = (const float*)d_in[0];
  const float* B = (const float*)d_in[1];
  float* out = (float*)d_out;
  const int N = in_sizes[0] / D_DIM;   // 8192
  const int M = in_sizes[1] / D_DIM;   // 8192
  dim3 grid(M / TILE_M, N / TILE_N);   // (128, 64)
  rbf_wmma_kernel<<<grid, 256, 0, stream>>>(A, B, out, N, M);
}
